// CVTGAD_18743237280504
// MI455X (gfx1250) — matrix operand order repaired and checked by reading, not compiled, not run on hardware
//
#include <hip/hip_runtime.h>
#include <hip/hip_bf16.h>

// ---------------------------------------------------------------------------
// CVTGAD fused loss for MI455X (gfx1250, wave32, WMMA).
//   node:  512 x [128x128x256] sim GEMM + row LSE   (f16 WMMA, f32 accum)
//   graph: [512x512x256] sim GEMM + online row LSE
//   final: std-weighted sum of means -> scalar
// ---------------------------------------------------------------------------

typedef __attribute__((ext_vector_type(16))) _Float16 v16h;
typedef __attribute__((ext_vector_type(8)))  float    v8f;

#define LDS_STRIDE 264   // 256 halves + 8 pad -> 528B row = 132 dwords (bank shift 4)
#define INV_TAU    2.0f  // 1 / 0.5
#define NB         512   // graphs
#define NPER       128   // nodes per graph
#define DDIM       256   // hidden dim

// --- load a 128x256 f32 tile, L2-normalize rows, store f16 row-major to LDS ---
__device__ inline void load_norm_tile(const float* __restrict__ src, _Float16* dst,
                                      int wave, int lane)
{
  for (int row = wave * 16; row < wave * 16 + 16; ++row) {
    const float4* p = (const float4*)(src + (size_t)row * DDIM + lane * 8);
    float4 x = p[0];
    float4 y = p[1];
    float ss = x.x * x.x + x.y * x.y + x.z * x.z + x.w * x.w
             + y.x * y.x + y.y * y.y + y.z * y.z + y.w * y.w;
#pragma unroll
    for (int m = 1; m < 32; m <<= 1) ss += __shfl_xor(ss, m, 32);
    float invn = 1.0f / fmaxf(sqrtf(ss), 1e-8f);
    union { _Float16 h[8]; uint4 q; } pk;
    pk.h[0] = (_Float16)(x.x * invn); pk.h[1] = (_Float16)(x.y * invn);
    pk.h[2] = (_Float16)(x.z * invn); pk.h[3] = (_Float16)(x.w * invn);
    pk.h[4] = (_Float16)(y.x * invn); pk.h[5] = (_Float16)(y.y * invn);
    pk.h[6] = (_Float16)(y.z * invn); pk.h[7] = (_Float16)(y.w * invn);
    *(uint4*)(dst + (size_t)row * LDS_STRIDE + lane * 8) = pk.q;
  }
}

// --- A fragment: 16x32 f16, rows rowBase..rowBase+15, K chunk k0 (ISA 7.12.2) ---
__device__ inline v16h load_a_frag(const _Float16* mat, int rowBase, int k0, int lane)
{
  const int mrow = rowBase + (lane & 15);
  const int khi  = (lane >> 4) << 3;             // +8 for hi half-lanes
  const _Float16* base = mat + mrow * LDS_STRIDE + k0 + khi;
  union { v16h v; unsigned u[8]; } f;
#pragma unroll
  for (int i = 0; i < 8; ++i) {
    const int k = ((i & 4) << 2) + ((i & 3) << 1); // vgpr i -> K offset
    f.u[i] = *(const unsigned*)(base + k);
  }
  return f.v;
}

// --- B fragment: 32x16 f16 = rows of S (sim = F * S^T), cols colBase.. ---
__device__ inline v16h load_b_frag(const _Float16* mat, int colBase, int k0, int lane)
{
  const int ncol = colBase + (lane & 15);
  const int kb   = k0 + ((lane >> 4) << 3);      // +8 for hi half-lanes
  const _Float16* base = mat + ncol * LDS_STRIDE + kb;  // 16B aligned
  union { v16h v; uint4 q[2]; } f;
  f.q[0] = *(const uint4*)(base);
  f.q[1] = *(const uint4*)(base + 16);
  return f.v;
}

// --- one wave computes a 16x128 strip of the 128x128 sim (raw dot products) ---
__device__ inline void gemm_strip(const _Float16* A, const _Float16* Bm,
                                  int wave, int lane, v8f acc[8])
{
#pragma unroll
  for (int kc = 0; kc < 8; ++kc) {
    v16h a = load_a_frag(A, wave * 16, kc * 32, lane);
#pragma unroll
    for (int t = 0; t < 8; ++t) {
      v16h b = load_b_frag(Bm, t * 16, kc * 32, lane);
      acc[t] = __builtin_amdgcn_wmma_f32_16x16x32_f16(
          false, a, false, b, (short)0, acc[t], false, false);
    }
  }
}

// row stats for row (hi*8 + r) of the strip: max and sum of exp over 128 cols
__device__ inline void row_stats(const v8f acc[8], int lane, int r,
                                 float& mx, float& s)
{
  mx = acc[0][r];
#pragma unroll
  for (int t = 1; t < 8; ++t) mx = fmaxf(mx, acc[t][r]);
#pragma unroll
  for (int m = 1; m < 16; m <<= 1) mx = fmaxf(mx, __shfl_xor(mx, m, 32));
  s = 0.0f;
#pragma unroll
  for (int t = 0; t < 8; ++t) s += expf(INV_TAU * (acc[t][r] - mx));
#pragma unroll
  for (int m = 1; m < 16; m <<= 1) s += __shfl_xor(s, m, 32);
}

// diagonal element of row (hi*8+r): tile t==wave, lane (l&15)==hi*8+r
__device__ inline float diag_val(const v8f acc[8], int wave, int lane, int r)
{
  const int hi = lane >> 4;
  float d = 0.0f;
#pragma unroll
  for (int t = 0; t < 8; ++t)
    if (t == wave) d = acc[t][r];
  return __shfl(d, hi * 24 + r, 32);
}

// ---------------------------------------------------------------------------
// Kernel 1: per-graph node-level InfoNCE. grid = 512, block = 256 (8 waves).
// ---------------------------------------------------------------------------
__global__ __launch_bounds__(256) void cvtgad_node_kernel(
    const float* __restrict__ h_f, const float* __restrict__ h_s,
    float* __restrict__ l_node)
{
  extern __shared__ char smem[];
  _Float16* F = (_Float16*)smem;
  _Float16* S = F + NPER * LDS_STRIDE;
  __shared__ float redbuf[16];

  const int wave = threadIdx.x >> 5;
  const int lane = threadIdx.x & 31;
  const int hi   = lane >> 4;
  const size_t gbase = (size_t)blockIdx.x * NPER * DDIM;

  load_norm_tile(h_f + gbase, F, wave, lane);
  load_norm_tile(h_s + gbase, S, wave, lane);
  __syncthreads();

  v8f acc[8] = {};
  gemm_strip(F, S, wave, lane, acc);

  float lossAcc = 0.0f;
#pragma unroll
  for (int r = 0; r < 8; ++r) {
    float mx, s;
    row_stats(acc, lane, r, mx, s);
    float pos = diag_val(acc, wave, lane, r) * INV_TAU;
    lossAcc += INV_TAU * mx + logf(s) - pos;   // -(pos - lse)
  }
  if ((lane & 15) == 0) redbuf[wave * 2 + hi] = lossAcc;
  __syncthreads();
  if (threadIdx.x == 0) {
    float tot = 0.0f;
#pragma unroll
    for (int i = 0; i < 16; ++i) tot += redbuf[i];
    l_node[blockIdx.x] = tot * (1.0f / (float)NPER);
  }
}

// ---------------------------------------------------------------------------
// Kernel 2: graph-level InfoNCE over pooled embeddings (512x512 sim).
// grid = 4 (128-row strips), block = 256; streams columns with online LSE.
// ---------------------------------------------------------------------------
__global__ __launch_bounds__(256) void cvtgad_graph_kernel(
    const float* __restrict__ hff, const float* __restrict__ hsf,
    float* __restrict__ l_graph)
{
  extern __shared__ char smem[];
  _Float16* A  = (_Float16*)smem;
  _Float16* Bm = A + NPER * LDS_STRIDE;

  const int wave = threadIdx.x >> 5;
  const int lane = threadIdx.x & 31;
  const int hi   = lane >> 4;
  const int R0   = blockIdx.x * 128;

  load_norm_tile(hff + (size_t)R0 * DDIM, A, wave, lane);

  float rm[8], rs[8], pos[8];
#pragma unroll
  for (int r = 0; r < 8; ++r) { rm[r] = -3.0e38f; rs[r] = 0.0f; pos[r] = 0.0f; }

  for (int c = 0; c < 4; ++c) {
    __syncthreads();                     // previous chunk's Bm reads done
    load_norm_tile(hsf + (size_t)c * 128 * DDIM, Bm, wave, lane);
    __syncthreads();

    v8f acc[8] = {};
    gemm_strip(A, Bm, wave, lane, acc);

#pragma unroll
    for (int r = 0; r < 8; ++r) {
      float mx, s;
      row_stats(acc, lane, r, mx, s);
      if (c == blockIdx.x) pos[r] = diag_val(acc, wave, lane, r) * INV_TAU;
      float cm = INV_TAU * mx;
      float nm = fmaxf(rm[r], cm);
      rs[r] = rs[r] * expf(rm[r] - nm) + s * expf(cm - nm);
      rm[r] = nm;
    }
  }

  if ((lane & 15) == 0) {
#pragma unroll
    for (int r = 0; r < 8; ++r)
      l_graph[R0 + wave * 16 + hi * 8 + r] = rm[r] + logf(rs[r]) - pos[r];
  }
}

// ---------------------------------------------------------------------------
// Kernel 3: uncertainty-weighted combination -> scalar.
// ---------------------------------------------------------------------------
__global__ __launch_bounds__(512) void cvtgad_finalize_kernel(
    const float* __restrict__ l_node, const float* __restrict__ l_graph,
    float* __restrict__ out)
{
  const int tid  = threadIdx.x;
  const int wave = tid >> 5;
  const int lane = tid & 31;
  float a = l_node[tid];
  float b = l_graph[tid];
  float v0 = a, v1 = a * a, v2 = b, v3 = b * b;
#pragma unroll
  for (int m = 1; m < 32; m <<= 1) {
    v0 += __shfl_xor(v0, m, 32);
    v1 += __shfl_xor(v1, m, 32);
    v2 += __shfl_xor(v2, m, 32);
    v3 += __shfl_xor(v3, m, 32);
  }
  __shared__ float buf[16][4];
  if (lane == 0) { buf[wave][0] = v0; buf[wave][1] = v1;
                   buf[wave][2] = v2; buf[wave][3] = v3; }
  __syncthreads();
  if (tid == 0) {
    float s0 = 0, s1 = 0, s2 = 0, s3 = 0;
    for (int i = 0; i < 16; ++i) {
      s0 += buf[i][0]; s1 += buf[i][1]; s2 += buf[i][2]; s3 += buf[i][3];
    }
    const float invB = 1.0f / (float)NB;
    float mn = s0 * invB, mg = s2 * invB;
    float vn = fmaxf(s1 * invB - mn * mn, 0.0f);   // population std (ddof=0)
    float vg = fmaxf(s3 * invB - mg * mg, 0.0f);
    out[0] = (sqrtf(vn) + 1e-6f) * mn + (sqrtf(vg) + 1e-6f) * mg;
  }
}

// ---------------------------------------------------------------------------
extern "C" void kernel_launch(void* const* d_in, const int* in_sizes, int n_in,
                              void* d_out, int out_size, void* d_ws, size_t ws_size,
                              hipStream_t stream)
{
  const float* h_f_final = (const float*)d_in[0];   // [512, 256]
  const float* h_s_final = (const float*)d_in[1];   // [512, 256]
  const float* h_f       = (const float*)d_in[2];   // [65536, 256]
  const float* h_s       = (const float*)d_in[3];   // [65536, 256]
  // d_in[4] = batch (uniform repeat(arange(512),128)) -- not needed.

  float* l_node  = (float*)d_ws;          // [512]
  float* l_graph = l_node + NB;           // [512]

  const size_t shmem = (size_t)2 * NPER * LDS_STRIDE * sizeof(_Float16); // 132 KB

  cvtgad_node_kernel<<<NB, 256, shmem, stream>>>(h_f, h_s, l_node);
  cvtgad_graph_kernel<<<4, 256, shmem, stream>>>(h_f_final, h_s_final, l_graph);
  cvtgad_finalize_kernel<<<1, 512, 0, stream>>>(l_node, l_graph, (float*)d_out);
}
